// Attention_47330539602252
// MI455X (gfx1250) — compile-verified, matrix operand order checked by reading, not run on hardware
//
#include <hip/hip_runtime.h>
#include <hip/hip_bf16.h>

// ---------------------------------------------------------------------------
// CDNA5 (gfx1250) fused Quad2-attention.
//   * all matmuls via v_wmma_f32_16x16x32_f16 (f32 accumulate)
//   * K/V tiles staged to LDS with global_load_async_to_lds_b128 (ASYNCcnt),
//     double-buffered, shared by all 8 waves of a workgroup
// ---------------------------------------------------------------------------

typedef __attribute__((ext_vector_type(16))) _Float16 v16h;
typedef __attribute__((ext_vector_type(8)))  float    v8f;

union FragU {
    v16h      h;
    uint4     q[2];
    _Float16  e[16];
};

// A-fragment (16x32 f16, row-major source) loader. The identical pattern serves
// as the B-fragment of the transpose of a row-major matrix (lane roles swap).
// Lane L: row = L&15 ; halves e[0..7] = cols c..c+7, e[8..15] = cols c+16..c+23
// with c = 8*(L>>4).  Two 16-byte loads (global_load_b128 or ds_load_b128).
__device__ __forceinline__ v16h load_frag_rm(const _Float16* tile, int stride, int lane) {
    const int r = lane & 15;
    const int c = (lane >> 4) << 3;          // 0 or 8
    const _Float16* p = tile + (size_t)r * stride + c;
    FragU f;
    f.q[0] = *(const uint4*)(p);
    f.q[1] = *(const uint4*)(p + 16);
    return f.h;
}

__device__ __forceinline__ v8f wmma16(v16h a, v16h b, v8f c) {
    // D = A(16x32 f16) * B(32x16 f16) + C(16x16 f32)
    return __builtin_amdgcn_wmma_f32_16x16x32_f16(false, a, false, b,
                                                  (short)0, c, false, false);
}

// ---------------------------------------------------------------------------
// 0) f32 -> f16 conversion (memory-trivial: ~30 MB total)
// ---------------------------------------------------------------------------
__global__ void cvt_f32_f16(const float* __restrict__ in, _Float16* __restrict__ out, int n) {
    int i = blockIdx.x * blockDim.x + threadIdx.x;
    int stride = gridDim.x * blockDim.x;
    for (; i < n; i += stride) out[i] = (_Float16)in[i];
}

// ---------------------------------------------------------------------------
// 1) QKV GEMM:  qkv[m, j] = sum_k xh[m,k] * wh[j,k],  M=8192, J=2304, K=768
//    Epilogue scatters to q[bh,n,d], k[bh,n,d], vt[bh,d,n] (f16).
//    Wave tile 32x64 (8 wmma / k-step), block = 8 waves => 128x128 tile.
// ---------------------------------------------------------------------------
__global__ void __launch_bounds__(256)
qkv_gemm(const _Float16* __restrict__ xh, const _Float16* __restrict__ wh,
         _Float16* __restrict__ qh, _Float16* __restrict__ kh, _Float16* __restrict__ vth)
{
    const int lane = threadIdx.x & 31;
    const int wave = threadIdx.x >> 5;
    const int row0 = blockIdx.x * 128 + (wave & 3) * 32;
    const int col0 = blockIdx.y * 128 + (wave >> 2) * 64;

    v8f acc[2][4] = {};
    for (int k0 = 0; k0 < 768; k0 += 32) {
        v16h a0 = load_frag_rm(xh + (size_t)row0        * 768 + k0, 768, lane);
        v16h a1 = load_frag_rm(xh + (size_t)(row0 + 16) * 768 + k0, 768, lane);
#pragma unroll
        for (int j = 0; j < 4; ++j) {
            v16h b = load_frag_rm(wh + (size_t)(col0 + j * 16) * 768 + k0, 768, lane);
            acc[0][j] = wmma16(a0, b, acc[0][j]);
            acc[1][j] = wmma16(a1, b, acc[1][j]);
        }
    }

    const int lr = lane & 15, hi = lane >> 4;
#pragma unroll
    for (int i = 0; i < 2; ++i) {
#pragma unroll
        for (int j = 0; j < 4; ++j) {
            const int colb  = col0 + j * 16;          // multiple of 16
            const int s     = colb / 768;             // 0:q 1:k 2:v (tile never straddles)
            const int rem   = colb - s * 768;
            const int h     = rem >> 6;
            const int d     = (rem & 63) + lr;
#pragma unroll
            for (int r = 0; r < 8; ++r) {
                const int m = row0 + i * 16 + r + 8 * hi;     // D-tile row mapping
                const int b = m >> 10, n = m & 1023;
                const size_t bh = (size_t)(b * 12 + h);
                const _Float16 val = (_Float16)acc[i][j][r];
                if (s == 0)      qh [(bh * 1024 + n) * 64 + d]  = val;
                else if (s == 1) kh [(bh * 1024 + n) * 64 + d]  = val;
                else             vth[(bh * 64 + d) * 1024 + n]  = val;   // V transposed
            }
        }
    }
}

// ---------------------------------------------------------------------------
// 2) Fused Quad2 attention.
//    One block = 8 waves, all on the SAME (b,h); wave w owns the 16-row Q block
//    (blockIdx.x&7)*8 + w.  K (32x64) and V^T (64x32) chunks are staged to LDS
//    with global_load_async_to_lds_b128 (one b128 per thread per tile),
//    double-buffered on ASYNCcnt, so K/V is read from HBM/L2 once per block
//    instead of once per wave (8x traffic reduction).
//    Single pass (Quad2 terms are non-negative => no rescaling):
//      S  = (Q*0.125) K^T ; T = (S+5)^2 (f16, via per-wave LDS transpose)
//      O += T V ; rowsum += T * ones    (ones-WMMA: same D mapping as O)
// ---------------------------------------------------------------------------
__device__ __forceinline__ void stage_kv_async(const _Float16* __restrict__ kp,
                                               const _Float16* __restrict__ vp,
                                               _Float16* kdst, _Float16* vdst,
                                               int m0, int t)
{
    // K tile: rows m0..m0+31 (stride 64), 2048 halves; thread t moves 8 halves.
    uint32_t koff = (uint32_t)(uintptr_t)(kdst + t * 8);          // LDS byte offset
    const _Float16* kg = kp + (size_t)(m0 + (t >> 3)) * 64 + ((t & 7) << 3);
    asm volatile("global_load_async_to_lds_b128 %0, %1, off"
                 :: "v"(koff), "v"(kg) : "memory");
    // V^T tile: rows d=0..63 (stride 1024), cols m0..m0+31 -> LDS [64][32].
    uint32_t voff = (uint32_t)(uintptr_t)(vdst + t * 8);
    const _Float16* vg = vp + (size_t)(t >> 2) * 1024 + m0 + ((t & 3) << 3);
    asm volatile("global_load_async_to_lds_b128 %0, %1, off"
                 :: "v"(voff), "v"(vg) : "memory");
}

__global__ void __launch_bounds__(256)
attn_quad(const _Float16* __restrict__ qh, const _Float16* __restrict__ kh,
          const _Float16* __restrict__ vth, _Float16* __restrict__ aoh)
{
    __shared__ _Float16 kbuf[2][32 * 64];    // 2 x 4 KB  K chunk
    __shared__ _Float16 vbuf[2][64 * 32];    // 2 x 4 KB  V^T chunk
    __shared__ _Float16 tbuf[8][16 * 32];    // 8 x 1 KB  per-wave T transpose

    const int t    = threadIdx.x;
    const int lane = t & 31;
    const int wave = t >> 5;
    const int bh   = blockIdx.x >> 3;                 // 96 (b,h) pairs
    const int qb   = (blockIdx.x & 7) * 8 + wave;     // 64 q-blocks per (b,h)

    const _Float16* qp = qh  + (size_t)bh * 1024 * 64 + (size_t)qb * 16 * 64;
    const _Float16* kp = kh  + (size_t)bh * 1024 * 64;
    const _Float16* vp = vth + (size_t)bh * 64 * 1024;

    // Q fragments, scale 1/sqrt(64) = 0.125 folded in (exact in f16)
    FragU aq0, aq1, ones;
    aq0.h = load_frag_rm(qp,      64, lane);
    aq1.h = load_frag_rm(qp + 32, 64, lane);
#pragma unroll
    for (int e = 0; e < 16; ++e) {
        aq0.e[e]  = aq0.e[e] * (_Float16)0.125f;
        aq1.e[e]  = aq1.e[e] * (_Float16)0.125f;
        ones.e[e] = (_Float16)1.0f;
    }

    v8f o[4] = {};
    v8f rs   = {};
    _Float16* tl = tbuf[wave];
    const int lr = lane & 15, hi = lane >> 4;

    // prologue: stage chunk 0
    stage_kv_async(kp, vp, kbuf[0], vbuf[0], 0, t);

    for (int c = 0; c < 32; ++c) {
        if (c + 1 < 32) {
            // buffer (c+1)&1 was last read in chunk c-1; the trailing barrier of
            // iteration c-1 guarantees every wave is done with it.
            stage_kv_async(kp, vp, kbuf[(c + 1) & 1], vbuf[(c + 1) & 1],
                           (c + 1) * 32, t);
            asm volatile("s_wait_asynccnt 0x2" ::: "memory");   // chunk c done
        } else {
            asm volatile("s_wait_asynccnt 0x0" ::: "memory");
        }
        __syncthreads();                                        // publish chunk c

        const _Float16* kt = kbuf[c & 1];
        const _Float16* vt = vbuf[c & 1];

        // --- scores: two 16x16 S tiles over hd=64 (2 k-steps each) ---
        v8f s0 = {}, s1 = {};
        s0 = wmma16(aq0.h, load_frag_rm(kt,           64, lane), s0);
        s0 = wmma16(aq1.h, load_frag_rm(kt + 32,      64, lane), s0);
        s1 = wmma16(aq0.h, load_frag_rm(kt + 16 * 64, 64, lane), s1);
        s1 = wmma16(aq1.h, load_frag_rm(kt + 16 * 64 + 32, 64, lane), s1);

        // --- Quad2: t = (s + 5)^2 ; D-layout -> row-major 16x32 in LDS ---
#pragma unroll
        for (int r = 0; r < 8; ++r) {
            const int row = r + 8 * hi;
            float t0 = s0[r] + 5.0f; t0 *= t0;
            float t1 = s1[r] + 5.0f; t1 *= t1;
            tl[row * 32 + lr]      = (_Float16)t0;
            tl[row * 32 + 16 + lr] = (_Float16)t1;
        }
        v16h at = load_frag_rm(tl, 32, lane);    // re-read as A fragment

        // --- accumulate O += T*V and rowsum += T*1 ---
#pragma unroll
        for (int j = 0; j < 4; ++j)
            o[j] = wmma16(at, load_frag_rm(vt + (j * 16) * 32, 32, lane), o[j]);
        rs = wmma16(at, ones.h, rs);

        __syncthreads();                         // retire chunk c before reuse
    }

    // rowsum D-tile columns are all identical => rs[r] matches o[j][r]'s row.
    const int b = bh / 12, h = bh - (bh / 12) * 12;
#pragma unroll
    for (int j = 0; j < 4; ++j) {
#pragma unroll
        for (int r = 0; r < 8; ++r) {
            const int n = qb * 16 + r + 8 * hi;
            const int cc = h * 64 + j * 16 + lr;
            aoh[((size_t)b * 1024 + n) * 768 + cc] = (_Float16)(o[j][r] / rs[r]);
        }
    }
}

// ---------------------------------------------------------------------------
// 3) Output projection: out[m,c] = sum_k aoh[m,k]*pw[c,k] + bias[c]  (f32 out)
// ---------------------------------------------------------------------------
__global__ void __launch_bounds__(256)
proj_gemm(const _Float16* __restrict__ aoh, const _Float16* __restrict__ pwh,
          const float* __restrict__ bias, float* __restrict__ out)
{
    const int lane = threadIdx.x & 31;
    const int wave = threadIdx.x >> 5;
    const int row0 = blockIdx.x * 128 + (wave & 3) * 32;
    const int col0 = blockIdx.y * 128 + (wave >> 2) * 64;

    v8f acc[2][4] = {};
    for (int k0 = 0; k0 < 768; k0 += 32) {
        v16h a0 = load_frag_rm(aoh + (size_t)row0        * 768 + k0, 768, lane);
        v16h a1 = load_frag_rm(aoh + (size_t)(row0 + 16) * 768 + k0, 768, lane);
#pragma unroll
        for (int j = 0; j < 4; ++j) {
            v16h b = load_frag_rm(pwh + (size_t)(col0 + j * 16) * 768 + k0, 768, lane);
            acc[0][j] = wmma16(a0, b, acc[0][j]);
            acc[1][j] = wmma16(a1, b, acc[1][j]);
        }
    }

    const int lr = lane & 15, hi = lane >> 4;
#pragma unroll
    for (int i = 0; i < 2; ++i) {
#pragma unroll
        for (int j = 0; j < 4; ++j) {
            const int c  = col0 + j * 16 + lr;
            const float bv = bias[c];
#pragma unroll
            for (int r = 0; r < 8; ++r) {
                const int m = row0 + i * 16 + r + 8 * hi;
                out[(size_t)m * 768 + c] = acc[i][j][r] + bv;
            }
        }
    }
}

// ---------------------------------------------------------------------------
// Launch
// ---------------------------------------------------------------------------
extern "C" void kernel_launch(void* const* d_in, const int* in_sizes, int n_in,
                              void* d_out, int out_size, void* d_ws, size_t ws_size,
                              hipStream_t stream)
{
    const float* x      = (const float*)d_in[0];   // [8,1024,768]
    const float* qkv_w  = (const float*)d_in[1];   // [2304,768]
    const float* proj_w = (const float*)d_in[2];   // [768,768]
    const float* proj_b = (const float*)d_in[3];   // [768]
    float* out = (float*)d_out;                    // [8,1024,768]

    constexpr int XN   = 8 * 1024 * 768;   // 6291456
    constexpr int WQN  = 2304 * 768;       // 1769472
    constexpr int WPN  = 768 * 768;        // 589824

    _Float16* ws   = (_Float16*)d_ws;
    _Float16* xh   = ws;
    _Float16* wqkv = xh   + XN;
    _Float16* wprj = wqkv + WQN;
    _Float16* qh   = wprj + WPN;
    _Float16* kh   = qh   + XN;            // [96,1024,64]
    _Float16* vth  = kh   + XN;            // [96,64,1024]  (V transposed)
    _Float16* aoh  = vth  + XN;            // [8,1024,768]  attention output, f16
    // total ws use: ~68 MB

    cvt_f32_f16<<<dim3(4096), dim3(256), 0, stream>>>(x,      xh,   XN);
    cvt_f32_f16<<<dim3(2048), dim3(256), 0, stream>>>(qkv_w,  wqkv, WQN);
    cvt_f32_f16<<<dim3(1024), dim3(256), 0, stream>>>(proj_w, wprj, WPN);

    qkv_gemm <<<dim3(64, 18), dim3(256), 0, stream>>>(xh, wqkv, qh, kh, vth);
    attn_quad<<<dim3(768),    dim3(256), 0, stream>>>(qh, kh, vth, aoh);
    proj_gemm<<<dim3(64, 6),  dim3(256), 0, stream>>>(aoh, wprj, proj_b, out);
}
